// LoopedMLP_62646392979735
// MI455X (gfx1250) — compile-verified, matrix operand order checked by reading, not run on hardware
//
#include <hip/hip_runtime.h>

typedef __bf16 bf16;
typedef __attribute__((ext_vector_type(16))) __bf16 v16bf;
typedef __attribute__((ext_vector_type(8)))  __bf16 v8bf;
typedef __attribute__((ext_vector_type(4)))  __bf16 v4bf;
typedef __attribute__((ext_vector_type(8)))  float  v8f;

#define N_EMBD 1024
#define ITERSZ 4096
#define FULLSZ 12288
#define TOK    4096   // B*T
#define CHUNK  1024

// async global->LDS 16B copy (gfx1250); ldsa = LDS byte address, ga = 64-bit global addr
#define ASYNC_B128(ldsa, ga) \
  asm volatile("global_load_async_to_lds_b128 %0, %1, off" :: "v"(ldsa), "v"(ga) : "memory")
#define WAIT_ASYNC0() asm volatile("s_wait_asynccnt 0x0" ::: "memory")

// ---------------------------------------------------------------------------
// Tiled WMMA GEMM:  C[m][n] = act( sum_k A[m][k]*B[n][k] + bias[n] )
// A: MxK row-major (fp32 or bf16), B: NxK row-major bf16 (weight), C: MxN.
// Block = 256 threads = 8 waves (4 x 2); tile 128x128, K-step 32.
// B tile (and A tile when bf16) moved with GLOBAL_LOAD_ASYNC_TO_LDS_B128.
// ---------------------------------------------------------------------------
template<int AF32, int RELU, int OBF16>
__global__ __launch_bounds__(256)
void gemm_bt(const void* __restrict__ Ap, const bf16* __restrict__ Bp,
             const float* __restrict__ bias, void* __restrict__ Cp,
             int Kd, int lda, int ldb, int ldc)
{
  const int BKP = 40;                       // bf16 elems per LDS row (80B pitch, 16B aligned)
  __shared__ alignas(16) bf16 As[128 * BKP];
  __shared__ alignas(16) bf16 Bs[128 * BKP];
  const int row0 = blockIdx.y * 128, col0 = blockIdx.x * 128;
  const int tid = threadIdx.x, lane = tid & 31, wid = tid >> 5;
  const int wm = wid >> 1, wn = wid & 1;    // wave tile: 32 rows x 64 cols
  const int lr = lane & 15, hi = lane >> 4, lo = hi * 8;

  // ---- B tile: 2 x 16B async copies per thread per K-step ----
  const int rB = tid >> 2, cB = tid & 3;    // 64 rows per issue, 4 x 16B segs per row
  unsigned long long gB0 = (unsigned long long)(uintptr_t)Bp
                         + ((size_t)(col0 + rB) * ldb) * 2 + (size_t)cB * 16;
  unsigned long long gB1 = gB0 + (size_t)64 * ldb * 2;
  const unsigned lB0 = (unsigned)(uintptr_t)&Bs[rB * BKP + cB * 8];
  const unsigned lB1 = (unsigned)(uintptr_t)&Bs[(rB + 64) * BKP + cB * 8];

  // ---- A tile: fp32 path (load+convert) or bf16 path (async) ----
  const char* pA[4]; bf16* sA[4];           // fp32 path: 4 x float4 per thread
  unsigned long long gA0 = 0, gA1 = 0;      // bf16 path: 2 x 16B async per thread
  unsigned lA0 = 0, lA1 = 0;
  if (AF32) {
    const int rA = tid >> 3, cA = tid & 7;  // 32 rows per issue, 8 x 16B segs per row
    for (int i = 0; i < 4; ++i) {
      pA[i] = (const char*)Ap + ((size_t)(row0 + rA + 32 * i) * lda + cA * 4) * 4;
      sA[i] = &As[(rA + 32 * i) * BKP + cA * 4];
    }
  } else {
    const int rA = tid >> 2, cA = tid & 3;
    gA0 = (unsigned long long)(uintptr_t)Ap
        + ((size_t)(row0 + rA) * lda) * 2 + (size_t)cA * 16;
    gA1 = gA0 + (size_t)64 * lda * 2;
    lA0 = (unsigned)(uintptr_t)&As[rA * BKP + cA * 8];
    lA1 = (unsigned)(uintptr_t)&As[(rA + 64) * BKP + cA * 8];
  }

  v8f acc[2][4];
  for (int i = 0; i < 2; ++i)
    for (int j = 0; j < 4; ++j)
      for (int v = 0; v < 8; ++v) acc[i][j][v] = 0.0f;

  for (int k0 = 0; k0 < Kd; k0 += 32) {
    ASYNC_B128(lB0, gB0);
    ASYNC_B128(lB1, gB1);
    gB0 += 64; gB1 += 64;
    if (AF32) {
      for (int i = 0; i < 4; ++i) {
        float4 v = *(const float4*)pA[i];
        pA[i] += 128;
        v4bf b; b[0] = (bf16)v.x; b[1] = (bf16)v.y; b[2] = (bf16)v.z; b[3] = (bf16)v.w;
        *(v4bf*)sA[i] = b;
      }
    } else {
      ASYNC_B128(lA0, gA0);
      ASYNC_B128(lA1, gA1);
      gA0 += 64; gA1 += 64;
    }
    WAIT_ASYNC0();
    __syncthreads();

    v16bf afrag[2];
    for (int mi = 0; mi < 2; ++mi) {        // A frag per ISA 16-bit 16x32 layout
      const bf16* p = &As[(wm * 32 + mi * 16 + lr) * BKP + lo];
      v8bf x0 = *(const v8bf*)p;            // K = lo..lo+7
      v8bf x1 = *(const v8bf*)(p + 16);     // K = 16+lo..16+lo+7
      for (int q = 0; q < 8; ++q) { afrag[mi][q] = x0[q]; afrag[mi][8 + q] = x1[q]; }
    }
    for (int ni = 0; ni < 4; ++ni) {
      const bf16* p = &Bs[(wn * 64 + ni * 16 + lr) * BKP + lo];
      v8bf x0 = *(const v8bf*)p;
      v8bf x1 = *(const v8bf*)(p + 16);
      v16bf bfrag;
      for (int q = 0; q < 8; ++q) { bfrag[q] = x0[q]; bfrag[8 + q] = x1[q]; }
      for (int mi = 0; mi < 2; ++mi)
        acc[mi][ni] = __builtin_amdgcn_wmma_f32_16x16x32_bf16(
            false, afrag[mi], false, bfrag, (short)0, acc[mi][ni], false, false);
    }
    __syncthreads();
  }

  // Epilogue: C 16x16 f32 layout -> lane<16: N=lane, M=v; lane>=16: M=8+v
  for (int mi = 0; mi < 2; ++mi)
    for (int ni = 0; ni < 4; ++ni) {
      int gn = col0 + wn * 64 + ni * 16 + lr;
      float bv = bias ? bias[gn] : 0.0f;
      for (int v = 0; v < 8; ++v) {
        int gm = row0 + wm * 32 + mi * 16 + (hi << 3) + v;
        float val = acc[mi][ni][v] + bv;
        if (RELU) val = fmaxf(val, 0.0f);
        if (OBF16) ((bf16*)Cp)[(size_t)gm * ldc + gn] = (bf16)val;
        else       ((float*)Cp)[(size_t)gm * ldc + gn] = val;
      }
    }
}

// ---------------------------------------------------------------------------
// Per-row softmax + exact K-th order statistic (4-pass radix select in LDS)
// and novelty-hit counting with stable-tie + straight-through fp32 "==1.0" test.
// One workgroup per token row of 12288 logits.
// ---------------------------------------------------------------------------
__global__ __launch_bounds__(256)
void select_rows(const float* __restrict__ logits, unsigned* __restrict__ flags)
{
  __shared__ unsigned u[FULLSZ];            // orderable-uint logits (48 KB)
  __shared__ unsigned hist[256];
  __shared__ float    fred[256];
  __shared__ unsigned ured[256];
  __shared__ unsigned sstat[2];
  const int tid = threadIdx.x;
  const float* rowp = logits + (size_t)blockIdx.x * FULLSZ;

  float lmax = -3.4e38f;
  for (int j = tid; j < FULLSZ; j += 256) {
    float f = rowp[j];
    unsigned b = __float_as_uint(f);
    u[j] = (b & 0x80000000u) ? ~b : (b | 0x80000000u);
    lmax = fmaxf(lmax, f);
  }
  fred[tid] = lmax; __syncthreads();
  for (int s = 128; s > 0; s >>= 1) { if (tid < s) fred[tid] = fmaxf(fred[tid], fred[tid + s]); __syncthreads(); }
  const float m = fred[0]; __syncthreads();

  float lsum = 0.0f;
  for (int j = tid; j < FULLSZ; j += 256) {
    unsigned s = u[j];
    unsigned b = (s & 0x80000000u) ? (s & 0x7fffffffu) : ~s;
    lsum += __expf(__uint_as_float(b) - m);
  }
  fred[tid] = lsum; __syncthreads();
  for (int s = 128; s > 0; s >>= 1) { if (tid < s) fred[tid] += fred[tid + s]; __syncthreads(); }
  const float invZ = 1.0f / fred[0]; __syncthreads();

  // radix select: tau = K-th largest orderable value, tieTake = ties to include
  unsigned prefix = 0, need = ITERSZ;
  for (int pass = 3; pass >= 0; --pass) {
    const int sh = pass * 8;
    hist[tid] = 0; __syncthreads();
    for (int j = tid; j < FULLSZ; j += 256) {
      unsigned v = u[j];
      unsigned hibits = (pass == 3) ? 0u : (v >> (sh + 8));
      if (hibits == prefix) atomicAdd(&hist[(v >> sh) & 255u], 1u);
    }
    __syncthreads();
    if (tid == 0) {
      unsigned cum = 0; int b = 255;
      for (;; --b) { unsigned c = hist[b]; if (cum + c >= need || b == 0) break; cum += c; }
      sstat[0] = (prefix << 8) | (unsigned)b;
      sstat[1] = need - cum;
    }
    __syncthreads();
    prefix = sstat[0]; need = sstat[1]; __syncthreads();
  }
  const unsigned tau = prefix, tieTake = need;

  unsigned lhit = 0, ltlo = 0;
  for (int j = tid; j < FULLSZ; j += 256) {
    unsigned v = u[j];
    if (v > tau) {
      if (j >= ITERSZ) {
        unsigned b = (v & 0x80000000u) ? (v & 0x7fffffffu) : ~v;
        float p  = __expf(__uint_as_float(b) - m) * invZ;
        float t2 = __fsub_rn(__fadd_rn(1.0f, p), p);   // straight-through fwd value
        if (t2 == 1.0f) lhit++;
      }
    } else if (v == tau && j < ITERSZ) ltlo++;         // ties take lowest indices first
  }
  ured[tid] = (lhit << 16) | ltlo; __syncthreads();
  for (int s = 128; s > 0; s >>= 1) { if (tid < s) ured[tid] += ured[tid + s]; __syncthreads(); }
  if (tid == 0) {
    unsigned hit = ured[0] >> 16, tlo = ured[0] & 0xffffu;
    unsigned b = (tau & 0x80000000u) ? (tau & 0x7fffffffu) : ~tau;
    float pt = __expf(__uint_as_float(b) - m) * invZ;
    float t2 = __fsub_rn(__fadd_rn(1.0f, pt), pt);
    unsigned extra = (tieTake > tlo && t2 == 1.0f) ? (tieTake - tlo) : 0u;
    atomicAdd(&flags[1], hit + extra);
  }
}

// ---------------------------------------------------------------------------
// Small helper kernels
// ---------------------------------------------------------------------------
__global__ void conv_f32_bf16(const float* __restrict__ s, bf16* __restrict__ d, int n) {
  int i = blockIdx.x * 256 + threadIdx.x;
  if (i < n) d[i] = (bf16)s[i];
}
// compact w_proj[:, :4096] -> bf16 [1024][4096]
__global__ void conv_wproj(const float* __restrict__ s, bf16* __restrict__ d) {
  int i = blockIdx.x * 256 + threadIdx.x;      // i < 1024*4096
  int dd = i >> 12, f = i & 4095;
  d[i] = (bf16)s[(size_t)dd * FULLSZ + f];
}
// c[d] = b_proj[d] + sum_{f>=4096} relu(b_mfc[f]) * w_proj[d][f]
__global__ void cvec_kernel(const float* __restrict__ wproj, const float* __restrict__ bmfc,
                            const float* __restrict__ bproj, float* __restrict__ cvec) {
  __shared__ float red[256];
  int d = blockIdx.x; float s = 0.0f;
  for (int f = ITERSZ + threadIdx.x; f < FULLSZ; f += 256)
    s += fmaxf(bmfc[f], 0.0f) * wproj[(size_t)d * FULLSZ + f];
  red[threadIdx.x] = s; __syncthreads();
  for (int t = 128; t > 0; t >>= 1) { if (threadIdx.x < t) red[threadIdx.x] += red[threadIdx.x + t]; __syncthreads(); }
  if (threadIdx.x == 0) cvec[d] = bproj[d] + red[0];
}
__global__ void init_flags(unsigned* f) { if (threadIdx.x == 0) { f[0] = 0u; f[1] = 0u; } }
__global__ void reset_hits(unsigned* f) { if (threadIdx.x == 0) f[1] = 0u; }
__global__ void commit_x(const unsigned* __restrict__ flags, const float* __restrict__ xnew,
                         float* __restrict__ xcur, int n) {
  int i = blockIdx.x * 256 + threadIdx.x;
  if (i < n && flags[0] == 0u) xcur[i] = xnew[i];   // x_out = where(done_old, xc, x_new)
}
__global__ void update_done(unsigned* f) {
  if (threadIdx.x == 0) {
    float novelty = (float)f[1] * (1.0f / (float)ITERSZ);
    if (novelty < 0.7f) f[0] = 1u;                  // done |= novelty < R
  }
}

// ---------------------------------------------------------------------------
extern "C" void kernel_launch(void* const* d_in, const int* in_sizes, int n_in,
                              void* d_out, int out_size, void* d_ws, size_t ws_size,
                              hipStream_t stream) {
  (void)in_sizes; (void)n_in; (void)ws_size;
  const float* x      = (const float*)d_in[0];
  const float* w_mfc  = (const float*)d_in[1];
  const float* b_mfc  = (const float*)d_in[2];
  const float* w_proj = (const float*)d_in[3];
  const float* b_proj = (const float*)d_in[4];
  const float* w_fc   = (const float*)d_in[5];
  const float* b_fc   = (const float*)d_in[6];
  const float* w_pc   = (const float*)d_in[7];
  const float* b_pc   = (const float*)d_in[8];

  char* ws = (char*)d_ws; size_t off = 0;
  auto alloc = [&](size_t bytes) -> char* {
    char* p = ws + off; off += (bytes + 255) & ~(size_t)255; return p;
  };
  bf16*  W1b    = (bf16*) alloc((size_t)ITERSZ * N_EMBD * 2);   // w_mfc[:4096]
  bf16*  W2b    = (bf16*) alloc((size_t)N_EMBD * ITERSZ * 2);   // w_proj[:,:4096]
  bf16*  W3b    = (bf16*) alloc((size_t)N_EMBD * N_EMBD * 2);   // w_fc_ck
  bf16*  W4b    = (bf16*) alloc((size_t)FULLSZ * N_EMBD * 2);   // w_proj_ck
  float* cvec   = (float*)alloc((size_t)N_EMBD * 4);
  float* xcur   = (float*)alloc((size_t)TOK * N_EMBD * 4);
  float* xnew   = (float*)alloc((size_t)TOK * N_EMBD * 4);
  bf16*  h      = (bf16*) alloc((size_t)TOK * ITERSZ * 2);
  bf16*  ck     = (bf16*) alloc((size_t)TOK * N_EMBD * 2);
  float* logits = (float*)alloc((size_t)CHUNK * FULLSZ * 4);
  unsigned* flags = (unsigned*)alloc(256);

  hipMemcpyAsync(xcur, x, (size_t)TOK * N_EMBD * 4, hipMemcpyDeviceToDevice, stream);

  conv_f32_bf16<<<(ITERSZ * N_EMBD) / 256, 256, 0, stream>>>(w_mfc, W1b, ITERSZ * N_EMBD);
  conv_wproj  <<<(N_EMBD * ITERSZ) / 256, 256, 0, stream>>>(w_proj, W2b);
  conv_f32_bf16<<<(N_EMBD * N_EMBD) / 256, 256, 0, stream>>>(w_fc, W3b, N_EMBD * N_EMBD);
  conv_f32_bf16<<<(FULLSZ * N_EMBD) / 256, 256, 0, stream>>>(w_pc, W4b, FULLSZ * N_EMBD);
  cvec_kernel<<<N_EMBD, 256, 0, stream>>>(w_proj, b_mfc, b_proj, cvec);
  init_flags<<<1, 32, 0, stream>>>(flags);

  for (int it = 0; it < 3; ++it) {
    // h = relu(xcur @ W1^T + b_mfc)          [4096 x 4096]
    gemm_bt<1,1,1><<<dim3(ITERSZ/128, TOK/128), 256, 0, stream>>>(
        xcur, W1b, b_mfc, h, N_EMBD, N_EMBD, N_EMBD, ITERSZ);
    // xnew = h @ W2^T + cvec                  [4096 x 1024]
    gemm_bt<0,0,0><<<dim3(N_EMBD/128, TOK/128), 256, 0, stream>>>(
        h, W2b, cvec, xnew, ITERSZ, ITERSZ, ITERSZ, N_EMBD);
    // ck = relu(xnew @ W3^T + b_fc)           [4096 x 1024]
    gemm_bt<1,1,1><<<dim3(N_EMBD/128, TOK/128), 256, 0, stream>>>(
        xnew, W3b, b_fc, ck, N_EMBD, N_EMBD, N_EMBD, N_EMBD);
    reset_hits<<<1, 32, 0, stream>>>(flags);
    // logits + novelty count, 1024-token chunks to bound workspace
    for (int c0 = 0; c0 < TOK; c0 += CHUNK) {
      gemm_bt<0,0,0><<<dim3(FULLSZ/128, CHUNK/128), 256, 0, stream>>>(
          ck + (size_t)c0 * N_EMBD, W4b, b_pc, logits, N_EMBD, N_EMBD, N_EMBD, FULLSZ);
      select_rows<<<CHUNK, 256, 0, stream>>>(logits, flags);
    }
    commit_x<<<(TOK * N_EMBD) / 256, 256, 0, stream>>>(flags, xnew, xcur, TOK * N_EMBD);
    update_done<<<1, 32, 0, stream>>>(flags);          // after commit: old-done semantics
  }

  hipMemcpyAsync(d_out, xcur, (size_t)out_size * sizeof(float),
                 hipMemcpyDeviceToDevice, stream);
}
